// CTLSTM_50010599194914
// MI455X (gfx1250) — matrix operand order repaired and checked by reading, not compiled, not run on hardware
//
#include <hip/hip_runtime.h>
#include <math.h>

#define H      1024
#define BATCH  32
#define SEQ    512
#define M_TOK  (BATCH * SEQ)                 // 16384 tokens
#define PLANE  ((size_t)M_TOK * H)           // elements per output plane
#define W_COLS (2 * H)                       // W_rec row stride (2H)
#define NGATE  5
#define MT     32                            // M-tile per wave (2 x 16 subtiles)
#define LDSROW (H + 8)                       // padded LDS row: 1032 elem (516 dw, 4-bank lane skew)
#define LDS_BYTES (MT * LDSROW * 2)          // 66,048 B

typedef __attribute__((ext_vector_type(16))) __bf16 v16bf;
typedef __attribute__((ext_vector_type(8)))  float  v8f;

union FragBF { uint4 u[2]; v16bf v; };

__device__ __forceinline__ unsigned short f32_to_bf16_rne(float f) {
    unsigned int u = __float_as_uint(f);
    unsigned int r = u + 0x7FFFu + ((u >> 16) & 1u);
    return (unsigned short)(r >> 16);
}

// ---- one-time f32 -> bf16 conversion of the embedding table ----
__global__ __launch_bounds__(256) void cvt_emb_kernel(const float* __restrict__ src,
                                                      unsigned short* __restrict__ dst,
                                                      int n) {
    int i = blockIdx.x * 256 + threadIdx.x;
    if (i < n) dst[i] = f32_to_bf16_rne(src[i]);
}

// ---- one-time f32 -> bf16 conversion of W_rec[:, :H] (first H columns) ----
__global__ __launch_bounds__(256) void cvt_w_kernel(const float* __restrict__ w,
                                                    unsigned short* __restrict__ dst,
                                                    int n) {
    int i = blockIdx.x * 256 + threadIdx.x;   // over 7H * H elements
    if (i < n) {
        int r = i >> 10;          // row in [0, 7H)
        int k = i & (H - 1);      // col in [0, H)
        dst[i] = f32_to_bf16_rne(w[(size_t)r * W_COLS + k]);
    }
}

// ---- fused gather + bf16 WMMA GEMM (5 live gate groups) + CT-LSTM epilogue ----
// Block: 8 waves, one 32-token mTile shared via async-staged LDS A-panel,
// each wave owns one 16-wide h-tile (hGroup*8 + wave).
__global__ __launch_bounds__(256) void ctlstm_fused_kernel(
    const unsigned short* __restrict__ embbf,   // [1001, H] bf16
    const unsigned short* __restrict__ wbf,     // [7H, H]   bf16
    const float* __restrict__ b_rec,            // [7H]
    const int*   __restrict__ ev,               // [B, T] event ids
    const float* __restrict__ dur,              // [B, T] durations
    float* __restrict__ out)                    // [5, T, B, H]
{
    extern __shared__ unsigned short ldsA[];    // [MT][LDSROW] bf16, base offset 0

    const int lane  = threadIdx.x & 31;
    const int wave  = threadIdx.x >> 5;
    const int mTile = (int)(blockIdx.x >> 3) * MT;                 // 32-token tile
    const int hBase = (((int)(blockIdx.x & 7)) * 8 + wave) << 4;   // 16-h tile
    const int n  = lane & 15;
    const int hi = lane >> 4;

    // ---- async-stage the gathered A panel (MT rows x K=H) into LDS ----
    {
        const int tid = threadIdx.x;
        const int r   = tid >> 3;           // LDS row 0..31
        const int c   = (tid & 7) * 8;      // 8-elem (16B) chunk within 64-elem stripe
        const int mTok = mTile + r;         // token = t*B + b
        const int evid = ev[(mTok & 31) * SEQ + (mTok >> 5)];
        const unsigned short* grow = embbf + (size_t)evid * H + c;
        unsigned ldsOff = (unsigned)(r * LDSROW + c) * 2u;   // LDS byte offset
#pragma unroll
        for (int kk = 0; kk < H; kk += 64) {
            const unsigned short* gp = grow + kk;
            unsigned lo = ldsOff + (unsigned)kk * 2u;
            asm volatile("global_load_async_to_lds_b128 %0, %1, off"
                         :: "v"(lo), "v"(gp)
                         : "memory");
        }
        asm volatile("s_wait_asynccnt 0" ::: "memory");
        __syncthreads();
    }

    // live gate groups: gi, gz, go, gib, gd  (gf/gfb are dead in the reference)
    const int gateRow[NGATE] = {0, 2 * H, 3 * H, 4 * H, 6 * H};

    // ---- B setup: lane feeds B-column n = W-row gateRow[g] + hBase + n ----
    const unsigned short* brow[NGATE];
#pragma unroll
    for (int g = 0; g < NGATE; ++g)
        brow[g] = wbf + (size_t)(gateRow[g] + hBase + n) * H + hi * 8;

    v8f acc0[NGATE] = {};
    v8f acc1[NGATE] = {};

#pragma unroll 2
    for (int k = 0; k < H; k += 32) {
        // A frags from LDS: subtile rows n and 16+n, K chunks [hi*8,+8) and [16+hi*8,+8)
        FragBF a0, a1;
        a0.u[0] = *reinterpret_cast<const uint4*>(&ldsA[n * LDSROW + k + hi * 8]);
        a0.u[1] = *reinterpret_cast<const uint4*>(&ldsA[n * LDSROW + k + 16 + hi * 8]);
        a1.u[0] = *reinterpret_cast<const uint4*>(&ldsA[(16 + n) * LDSROW + k + hi * 8]);
        a1.u[1] = *reinterpret_cast<const uint4*>(&ldsA[(16 + n) * LDSROW + k + 16 + hi * 8]);
#pragma unroll
        for (int g = 0; g < NGATE; ++g) {
            // B frag: K chunks [hi*8,+8) and [16+hi*8,+8), reused by 2 WMMAs
            FragBF b;
            const uint4* p = reinterpret_cast<const uint4*>(brow[g] + k);
            b.u[0] = p[0];
            b.u[1] = p[2];
            acc0[g] = __builtin_amdgcn_wmma_f32_16x16x32_bf16(
                false, a0.v, false, b.v, (short)0, acc0[g], false, false);
            acc1[g] = __builtin_amdgcn_wmma_f32_16x16x32_bf16(
                false, a1.v, false, b.v, (short)0, acc1[g], false, false);
        }
    }

    // ---- epilogue ----
    float bias[NGATE];
#pragma unroll
    for (int g = 0; g < NGATE; ++g)
        bias[g] = b_rec[gateRow[g] + hBase + n];

    const size_t outCol = (size_t)hBase + n;
#pragma unroll
    for (int sub = 0; sub < 2; ++sub) {
        const v8f* acc = sub ? acc1 : acc0;
#pragma unroll
        for (int v = 0; v < 8; ++v) {
            const int m = mTile + sub * 16 + v + 8 * hi;   // C/D row for this VGPR
            const float d = dur[(m & 31) * SEQ + (m >> 5)];

            const float gi  = acc[0][v] + bias[0];
            const float gz  = acc[1][v] + bias[1];
            const float go  = acc[2][v] + bias[2];
            const float gib = acc[3][v] + bias[3];
            const float gd  = acc[4][v] + bias[4];

            const float si  = 1.0f / (1.0f + __expf(-gi));
            const float z   = tanhf(gz);
            const float so  = 1.0f / (1.0f + __expf(-go));
            const float sib = 1.0f / (1.0f + __expf(-gib));
            const float sd  = (gd > 20.0f) ? gd : log1pf(__expf(gd));  // softplus

            const float c   = si * z;
            const float cb  = sib * z;
            const float cd  = cb + (c - cb) * __expf(-sd * d);
            const float hd  = so * tanhf(cd);

            const size_t base = (size_t)m * H + outCol;
            out[base]              = hd;
            out[PLANE + base]      = c;
            out[2 * PLANE + base]  = cb;
            out[3 * PLANE + base]  = so;
            out[4 * PLANE + base]  = sd;
        }
    }
}

extern "C" void kernel_launch(void* const* d_in, const int* in_sizes, int n_in,
                              void* d_out, int out_size, void* d_ws, size_t ws_size,
                              hipStream_t stream) {
    (void)in_sizes; (void)n_in; (void)out_size; (void)ws_size;

    const int*   ev   = (const int*)d_in[0];    // event_seqs [B, T]
    const float* dur  = (const float*)d_in[1];  // duration_seqs [B, T]
    const float* emb  = (const float*)d_in[2];  // emb_table [1001, H]
    const float* wrec = (const float*)d_in[3];  // W_rec [7H, 2H]
    const float* brec = (const float*)d_in[4];  // b_rec [7H]
    float* out = (float*)d_out;

    // workspace: embbf (bf16) then wbf (bf16); both 16B-aligned
    unsigned short* embbf = (unsigned short*)d_ws;
    const int embElems = (1000 + 1) * H;               // 1,025,024 (x2 bytes, 16B aligned)
    unsigned short* wbf = embbf + embElems;
    const int wElems = 7 * H * H;                      // 7,340,032

    cvt_emb_kernel<<<(embElems + 255) / 256, 256, 0, stream>>>(emb, embbf, embElems);
    cvt_w_kernel<<<(wElems + 255) / 256, 256, 0, stream>>>(wrec, wbf, wElems);

    // 512 mTiles (32 tokens) * 8 hGroups; 8 waves/block, one 32x16 tile per wave
    ctlstm_fused_kernel<<<(M_TOK / MT) * 8, 256, LDS_BYTES, stream>>>(
        embbf, wbf, brec, ev, dur, out);
}
